// SparseConvCausalAttention_5669356831318
// MI455X (gfx1250) — matrix operand order repaired and checked by reading, not compiled, not run on hardware
//
#include <hip/hip_runtime.h>
#include <hip/hip_bf16.h>

// ---------------- problem constants ----------------
#define BB     4
#define HH     8
#define BHN    32          // B*H
#define DIM_   512
#define DH_    64
#define TT     1280        // padded seq (256 text + 1024 image)
#define TEXT   256
#define IMGL   1024
#define IMGW   32
#define NREAL  1279
#define QKV3   1536
#define KS_    5
#define PATCH  25
#define NEGINF (-3.0e38f)

// ---------------- workspace layout (f16 elements) ----------------
// all offsets are multiples of 2048 elems -> 16B-aligned vector loads safe
static constexpr size_t OFF_XH   = 0;                               // [B][T][DIM]        2,621,440
static constexpr size_t OFF_WQKV = OFF_XH   + (size_t)BB*TT*DIM_;   // [1536][512] (N,K)    786,432
static constexpr size_t OFF_WOUT = OFF_WQKV + (size_t)QKV3*DIM_;    // [512][512]  (N,K)    262,144
static constexpr size_t OFF_Q    = OFF_WOUT + (size_t)DIM_*DIM_;    // [BH][T][DH]        2,621,440
static constexpr size_t OFF_K    = OFF_Q    + (size_t)BHN*TT*DH_;
static constexpr size_t OFF_V    = OFF_K    + (size_t)BHN*TT*DH_;
static constexpr size_t OFF_VT   = OFF_V    + (size_t)BHN*TT*DH_;   // [BH][DH][T] (transposed v)
static constexpr size_t OFF_AO   = OFF_VT   + (size_t)BHN*DH_*TT;   // [B][T][DIM] head-merged attn out
// total = 16,777,216 f16 = 32 MB

// ---------------- WMMA fragment helpers ----------------
typedef __attribute__((ext_vector_type(16))) _Float16 v16h;
typedef __attribute__((ext_vector_type(8)))  _Float16 h8;
typedef __attribute__((ext_vector_type(8)))  float    v8f;

union AFrag { v16h v; h8 h[2]; _Float16 e[16]; };

__device__ __forceinline__ int lane_id() { return (int)(threadIdx.x & 31u); }

// A-matrix 16x32 f16, row-major source (stride in elems).
// lane<16: row=lane,    K = k0 + {0..7, 16..23}
// lane>=16: row=lane-16, K = k0 + {8..15, 24..31}
__device__ __forceinline__ v16h load_fragA(const _Float16* base, int stride, int row0, int k0) {
  int l  = lane_id();
  int r  = row0 + (l & 15);
  int kb = (l >> 4) << 3;
  const _Float16* p = base + (size_t)r * stride + k0 + kb;
  AFrag f;
  f.h[0] = *(const h8*)(p);
  f.h[1] = *(const h8*)(p + 16);
  return f.v;
}

// B-matrix 32x16 f16, source stored [col][k] (stride in elems).
// lane<16: col=lane,    K = k0 + {0..7, 16..23} ; lane>=16: col=lane-16, K = k0 + {8..15, 24..31}
__device__ __forceinline__ v16h load_fragB(const _Float16* base, int stride, int col0, int k0) {
  int l  = lane_id();
  int c  = col0 + (l & 15);
  int kb = (l >> 4) << 3;
  const _Float16* p = base + (size_t)c * stride + k0 + kb;
  AFrag f;
  f.h[0] = *(const h8*)(p);
  f.h[1] = *(const h8*)(p + 16);
  return f.v;
}

// A fragment gathered from an f32 row in LDS (softmax probabilities)
__device__ __forceinline__ v16h lds_fragA_f32(const float* s, int stride, int k0) {
  int l  = lane_id();
  int r  = l & 15;
  int kb = (l >> 4) << 3;
  const float* p = s + (size_t)r * stride + k0 + kb;
  AFrag f;
#pragma unroll
  for (int i = 0; i < 8; ++i) f.e[i]     = (_Float16)p[i];
#pragma unroll
  for (int i = 0; i < 8; ++i) f.e[8 + i] = (_Float16)p[16 + i];
  return f.v;
}

__device__ __forceinline__ v8f wmma16(v16h a, v16h b, v8f c) {
  return __builtin_amdgcn_wmma_f32_16x16x32_f16(false, a, false, b, (short)0, c, false, false);
}

// ---------------- kernel 1: fp32 -> f16 conversion + weight transpose ----------------
__global__ void prep_kernel(const float* __restrict__ x, const float* __restrict__ w_qkv,
                            const float* __restrict__ w_out, _Float16* __restrict__ ws) {
  _Float16* xh = ws + OFF_XH;
  _Float16* wq = ws + OFF_WQKV;
  _Float16* wo = ws + OFF_WOUT;
  const long XH_N = (long)BB * TT * DIM_;
  const long WQ_N = (long)QKV3 * DIM_;
  const long WO_N = (long)DIM_ * DIM_;
  const long total = XH_N + WQ_N + WO_N;
  for (long idx = (long)blockIdx.x * blockDim.x + threadIdx.x; idx < total;
       idx += (long)gridDim.x * blockDim.x) {
    if (idx < XH_N) {
      long b   = idx / ((long)TT * DIM_);
      long rem = idx % ((long)TT * DIM_);
      int  t   = (int)(rem / DIM_);
      int  d   = (int)(rem % DIM_);
      float v  = (t < NREAL) ? x[(b * NREAL + t) * DIM_ + d] : 0.0f;
      xh[idx] = (_Float16)v;
    } else if (idx < XH_N + WQ_N) {
      long i = idx - XH_N;
      int  n = (int)(i / DIM_);
      int  k = (int)(i % DIM_);
      wq[i] = (_Float16)w_qkv[(long)k * QKV3 + n];
    } else {
      long i = idx - XH_N - WQ_N;
      int  n = (int)(i / DIM_);
      int  k = (int)(i % DIM_);
      wo[i] = (_Float16)w_out[(long)k * DIM_ + n];
    }
  }
}

// ---------------- kernel 2: qkv = x @ w_qkv, split heads ----------------
// grid (80, 24), block 128 (4 waves); each wave: 16 rows x 64 cols
__global__ void __launch_bounds__(128) qkv_gemm_kernel(_Float16* __restrict__ ws) {
  const _Float16* xh = ws + OFF_XH;
  const _Float16* wq = ws + OFF_WQKV;
  _Float16* q_h = ws + OFF_Q;
  _Float16* k_h = ws + OFF_K;
  _Float16* v_h = ws + OFF_V;
  _Float16* v_t = ws + OFF_VT;

  int wave = (int)(threadIdx.x >> 5);
  int m0 = blockIdx.x * 64 + wave * 16;
  int n0 = blockIdx.y * 64;

  v8f acc[4] = {};
  for (int k0 = 0; k0 < DIM_; k0 += 32) {
    v16h a = load_fragA(xh, DIM_, m0, k0);
#pragma unroll
    for (int nt = 0; nt < 4; ++nt) {
      v16h bf = load_fragB(wq, DIM_, n0 + nt * 16, k0);
      acc[nt] = wmma16(a, bf, acc[nt]);
    }
  }

  int l   = lane_id();
  int col = l & 15;
  int rhi = (l >> 4) * 8;
#pragma unroll
  for (int nt = 0; nt < 4; ++nt) {
    int n   = n0 + nt * 16 + col;
    int sel = n >> 9;          // 0=q, 1=k, 2=v
    int nn  = n & 511;
    int hd  = nn >> 6;
    int dd  = nn & 63;
#pragma unroll
    for (int r = 0; r < 8; ++r) {
      int bt = m0 + r + rhi;
      int b  = bt / TT, t = bt % TT;
      int bh = b * HH + hd;
      long o = ((long)bh * TT + t) * DH_ + dd;
      float v = acc[nt][r];
      if (sel == 0)      q_h[o] = (_Float16)(v * 0.125f);   // q * DH^-0.5
      else if (sel == 1) k_h[o] = (_Float16)v;
      else {
        _Float16 hv = (_Float16)v;
        v_h[o] = hv;
        v_t[((long)bh * DH_ + dd) * TT + t] = hv;
      }
    }
  }
}

// ---------------- kernel 3: causal text attention ----------------
// 512 wave-tasks (32 bh x 16 row-tiles); block 64 = 2 waves; grid 256
#define TA_WAVES 2
__global__ void __launch_bounds__(64) text_attn_kernel(_Float16* __restrict__ ws) {
  __shared__ float sc[TA_WAVES][16][TEXT + 8];
  const _Float16* q_h = ws + OFF_Q;
  const _Float16* k_h = ws + OFF_K;
  const _Float16* v_t = ws + OFF_VT;
  _Float16* ao = ws + OFF_AO;

  int wave = (int)(threadIdx.x >> 5);
  int task = blockIdx.x * TA_WAVES + wave;
  int bh = task >> 4;
  int i0 = (task & 15) * 16;
  const _Float16* qb  = q_h + (long)bh * TT * DH_;
  const _Float16* kb  = k_h + (long)bh * TT * DH_;
  const _Float16* vtb = v_t + (long)bh * DH_ * TT;

  int l   = lane_id();
  int col = l & 15;
  int rhi = (l >> 4) * 8;

  // ---- scores = q_text @ k_text^T, causal mask ----
  v16h aq0 = load_fragA(qb, DH_, i0, 0);
  v16h aq1 = load_fragA(qb, DH_, i0, 32);
  for (int jt = 0; jt < 16; ++jt) {
    int j0 = jt * 16;
    v8f acc = {};
    acc = wmma16(aq0, load_fragB(kb, DH_, j0, 0), acc);
    acc = wmma16(aq1, load_fragB(kb, DH_, j0, 32), acc);
#pragma unroll
    for (int r = 0; r < 8; ++r) {
      int row = r + rhi;
      int ig  = i0 + row;
      int j   = j0 + col;
      sc[wave][row][j] = (j > ig) ? NEGINF : acc[r];
    }
  }
  __syncthreads();

  // ---- softmax (f32, per row) ----
  if (l < 16) {
    float* srow = sc[wave][l];
    float m = NEGINF;
    for (int j = 0; j < TEXT; ++j) m = fmaxf(m, srow[j]);
    float s = 0.0f;
    for (int j = 0; j < TEXT; ++j) { float e = __expf(srow[j] - m); srow[j] = e; s += e; }
    float inv = 1.0f / s;
    for (int j = 0; j < TEXT; ++j) srow[j] *= inv;
  }
  __syncthreads();

  // ---- out = attn @ v_text ----
  int b = bh >> 3, hd = bh & 7;
#pragma unroll
  for (int nt = 0; nt < 4; ++nt) {
    int d0 = nt * 16;
    v8f acc = {};
    for (int j0 = 0; j0 < TEXT; j0 += 32) {
      v16h a  = lds_fragA_f32(&sc[wave][0][0], TEXT + 8, j0);
      v16h bf = load_fragB(vtb, TT, d0, j0);
      acc = wmma16(a, bf, acc);
    }
#pragma unroll
    for (int r = 0; r < 8; ++r) {
      int row = r + rhi;
      int t = i0 + row;
      int d = d0 + col;
      ao[((long)b * TT + t) * DIM_ + hd * DH_ + d] = (_Float16)acc[r];
    }
  }
}

// ---------------- kernel 4: image attention (i2t via WMMA + 5x5 causal conv patch) ----
// 2048 wave-tasks (32 bh x 64 pixel-tiles); block 64 = 2 waves; grid 1024
#define IA_WAVES 2
#define NCOL 288            // 256 text + 25 patch, padded
__global__ void __launch_bounds__(64) img_attn_kernel(_Float16* __restrict__ ws,
                                                      const unsigned char* __restrict__ mask) {
  __shared__ float sc[IA_WAVES][16][NCOL];
  const _Float16* q_h = ws + OFF_Q;
  const _Float16* k_h = ws + OFF_K;
  const _Float16* v_h = ws + OFF_V;
  const _Float16* v_t = ws + OFF_VT;
  _Float16* ao = ws + OFF_AO;

  int wave = (int)(threadIdx.x >> 5);
  int task = blockIdx.x * IA_WAVES + wave;
  int bh  = task >> 6;
  int p0  = (task & 63) * 16;     // first pixel of tile
  int pi  = p0 >> 5;              // image row (fixed across tile)
  int pj0 = p0 & 31;              // first image col (0 or 16)
  int b = bh >> 3, hd = bh & 7;

  const _Float16* qb  = q_h + ((long)bh * TT + TEXT) * DH_;  // q_img rows indexed by pixel
  const _Float16* kb  = k_h + (long)bh * TT * DH_;
  const _Float16* vb  = v_h + (long)bh * TT * DH_;
  const _Float16* vtb = v_t + (long)bh * DH_ * TT;

  int l   = lane_id();
  int col = l & 15;
  int rhi = (l >> 4) * 8;

  // ---- image-to-text scores via WMMA ----
  v16h aq0 = load_fragA(qb, DH_, p0, 0);
  v16h aq1 = load_fragA(qb, DH_, p0, 32);
  for (int jt = 0; jt < 16; ++jt) {
    int j0 = jt * 16;
    v8f acc = {};
    acc = wmma16(aq0, load_fragB(kb, DH_, j0, 0), acc);
    acc = wmma16(aq1, load_fragB(kb, DH_, j0, 32), acc);
    int j = j0 + col;
    bool valid = mask[b * TEXT + j] != 0;
#pragma unroll
    for (int r = 0; r < 8; ++r)
      sc[wave][r + rhi][j] = valid ? acc[r] : NEGINF;
  }

  // ---- 5x5 causal conv patch scores (VALU; tiny) ----
  if (l < 16) {
    int pc = pj0 + l;
    const _Float16* qrow = qb + (long)(p0 + l) * DH_;
    float qr[DH_];
#pragma unroll
    for (int d = 0; d < DH_; ++d) qr[d] = (float)qrow[d];
    for (int e = 0; e < PATCH; ++e) {
      int kr = e / KS_, kc = e % KS_;
      int ky = pi + kr - 4, kx = pc + kc - 4;
      float v = NEGINF;
      if (ky >= 0 && kx >= 0) {
        const _Float16* krow = kb + (long)(TEXT + ky * IMGW + kx) * DH_;
        float s = 0.0f;
        for (int d = 0; d < DH_; ++d) s += qr[d] * (float)krow[d];
        v = s;
      }
      sc[wave][l][TEXT + e] = v;
    }
  }
  __syncthreads();

  // ---- joint softmax over 281 keys ----
  if (l < 16) {
    float* srow = sc[wave][l];
    float m = NEGINF;
    for (int j = 0; j < TEXT + PATCH; ++j) m = fmaxf(m, srow[j]);
    float s = 0.0f;
    for (int j = 0; j < TEXT + PATCH; ++j) { float e = __expf(srow[j] - m); srow[j] = e; s += e; }
    float inv = 1.0f / s;
    for (int j = 0; j < TEXT + PATCH; ++j) srow[j] *= inv;
  }
  __syncthreads();

  // ---- out = attn_i2t @ v_text (WMMA) + attn_patch . v_patch (VALU) ----
#pragma unroll
  for (int nt = 0; nt < 4; ++nt) {
    int d0 = nt * 16;
    v8f acc = {};
    for (int j0 = 0; j0 < TEXT; j0 += 32) {
      v16h a  = lds_fragA_f32(&sc[wave][0][0], NCOL, j0);
      v16h bf = load_fragB(vtb, TT, d0, j0);
      acc = wmma16(a, bf, acc);
    }
#pragma unroll
    for (int r = 0; r < 8; ++r) {
      int row = r + rhi;
      int p  = p0 + row;
      int pc = pj0 + row;
      int d  = d0 + col;
      float s = acc[r];
      for (int e = 0; e < PATCH; ++e) {
        int kr = e / KS_, kc = e % KS_;
        int ky = pi + kr - 4, kx = pc + kc - 4;
        if (ky >= 0 && kx >= 0)
          s += sc[wave][row][TEXT + e] * (float)vb[(long)(TEXT + ky * IMGW + kx) * DH_ + d];
      }
      int t = TEXT + p;
      ao[((long)b * TT + t) * DIM_ + hd * DH_ + d] = (_Float16)s;
    }
  }
}

// ---------------- kernel 5: out = attn_out @ w_out + b_out ----------------
// grid (80, 8), block 128 (4 waves); each wave 16 rows x 64 cols
__global__ void __launch_bounds__(128) out_proj_kernel(const _Float16* __restrict__ ws,
                                                       const float* __restrict__ b_out,
                                                       float* __restrict__ out) {
  const _Float16* ao = ws + OFF_AO;
  const _Float16* wo = ws + OFF_WOUT;
  int wave = (int)(threadIdx.x >> 5);
  int m0 = blockIdx.x * 64 + wave * 16;
  int n0 = blockIdx.y * 64;

  v8f acc[4] = {};
  for (int k0 = 0; k0 < DIM_; k0 += 32) {
    v16h a = load_fragA(ao, DIM_, m0, k0);
#pragma unroll
    for (int nt = 0; nt < 4; ++nt) {
      v16h bf = load_fragB(wo, DIM_, n0 + nt * 16, k0);
      acc[nt] = wmma16(a, bf, acc[nt]);
    }
  }

  int l   = lane_id();
  int col = l & 15;
  int rhi = (l >> 4) * 8;
#pragma unroll
  for (int nt = 0; nt < 4; ++nt) {
    int n = n0 + nt * 16 + col;
    float bias = b_out[n];
#pragma unroll
    for (int r = 0; r < 8; ++r) {
      int bt = m0 + r + rhi;
      int b = bt / TT, t = bt % TT;
      if (t < NREAL)
        out[((long)b * NREAL + t) * DIM_ + n] = acc[nt][r] + bias;
    }
  }
}

// ---------------- launch ----------------
extern "C" void kernel_launch(void* const* d_in, const int* in_sizes, int n_in,
                              void* d_out, int out_size, void* d_ws, size_t ws_size,
                              hipStream_t stream) {
  const float*         x     = (const float*)d_in[0];
  const unsigned char* mask  = (const unsigned char*)d_in[1];
  const float*         w_qkv = (const float*)d_in[2];
  const float*         w_out = (const float*)d_in[3];
  const float*         b_out = (const float*)d_in[4];
  float*     out = (float*)d_out;
  _Float16*  ws  = (_Float16*)d_ws;

  prep_kernel<<<2048, 256, 0, stream>>>(x, w_qkv, w_out, ws);

  dim3 gq(80, 24);
  qkv_gemm_kernel<<<gq, 128, 0, stream>>>(ws);

  text_attn_kernel<<<256, 64, 0, stream>>>(ws);

  img_attn_kernel<<<1024, 64, 0, stream>>>(ws, mask);

  dim3 go(80, 8);
  out_proj_kernel<<<go, 128, 0, stream>>>(ws, b_out, out);
}
